// Seq2SeqTransformer_77464030151112
// MI455X (gfx1250) — compile-verified, matrix operand order checked by reading, not run on hardware
//
#include <hip/hip_runtime.h>

#define TT 512
#define DD 512
#define HH 8
#define HDIM 64
#define BB 4
#define VV 32000
#define LL 6

typedef __bf16 bf16_t;
typedef __attribute__((ext_vector_type(16))) __bf16 v16bf;
typedef __attribute__((ext_vector_type(8)))  __bf16 v8bf;
typedef __attribute__((ext_vector_type(8)))  float  v8f;

union Frag16 { v16bf v; v8bf h[2]; };

__device__ __forceinline__ unsigned lds_addr32(const void* p) {
  // For LDS (shared) generic pointers the low 32 bits are the LDS byte address.
  return (unsigned)(unsigned long long)p;
}

__device__ __forceinline__ void async_ld_b128(unsigned lds, unsigned long long gaddr) {
  // GLOBAL_LOAD_ASYNC_TO_LDS_B128, GV mode (64-bit vaddr). Tracked by ASYNCcnt.
  asm volatile("global_load_async_to_lds_b128 %0, %1, off"
               :: "v"(lds), "v"(gaddr) : "memory");
}

__device__ __forceinline__ void wait_async(int n) {
  if (n == 0)
    asm volatile("s_wait_asynccnt 0x0" ::: "memory");
  else
    asm volatile("s_wait_asynccnt 0x3" ::: "memory");
}

// ---------------------------------------------------------------------------
// Generic bf16 WMMA GEMM:  C = alpha * A(MxK, row-major) * Bt(NxK, row-major)^T
// Double-buffered LDS staging via GLOBAL_LOAD_ASYNC_TO_LDS_B128 (ASYNCcnt).
// MODE 0: bf16 out (+bias, +relu)   MODE 1: f32 out (+bias)
// MODE 2: f32 out = resid + A*B + bias   (residual stream update, in place OK)
// Batched over grid.z with (outer, inner) stride decomposition via zdiv.
// ---------------------------------------------------------------------------
template<int WM, int WN, int MODE>
__global__ __launch_bounds__(256) void gemm_bf16_kernel(
    const bf16_t* __restrict__ A, int lda, long long sAo, long long sAi,
    const bf16_t* __restrict__ Bt, int ldb, long long sBo, long long sBi,
    void* __restrict__ Cv, int ldc, long long sCo, long long sCi,
    const float* __restrict__ resid,
    const float* __restrict__ bias,
    int zdiv, int K, float alpha, int relu)
{
  constexpr int BM = WM * 32;
  constexpr int BN = WN * 32;
  constexpr int NA = (BM * 4) / 256;   // 16B chunks per thread for A tile
  constexpr int NB = (BN * 4) / 256;   // 16B chunks per thread for B tile
  constexpr int LDP = 40;              // padded row pitch (elems): 80B, conflict-free

  __shared__ __align__(32) bf16_t As[2][BM * LDP];
  __shared__ __align__(32) bf16_t Bs[2][BN * LDP];

  const int tid  = threadIdx.x;
  const int wid  = tid >> 5;
  const int lane = tid & 31;
  const int wm   = wid / WN;
  const int wn   = wid % WN;
  const int zo   = blockIdx.z / zdiv;
  const int zi   = blockIdx.z % zdiv;
  const long long aoff = zo * sAo + zi * sAi;
  const long long boff = zo * sBo + zi * sBi;
  const long long coff = zo * sCo + zi * sCi;
  const int m0 = blockIdx.y * BM;
  const int n0 = blockIdx.x * BN;
  const int l16 = lane & 15;
  const int kh  = lane >> 4;

  // Per-thread staging chunk addresses (advance by 64B per K-step).
  unsigned long long ga[NA], gb[NB];
  unsigned la[2][NA], lb[2][NB];
#pragma unroll
  for (int j = 0; j < NA; ++j) {
    int i = tid + j * 256;
    int r = i >> 2, c = (i & 3) * 8;
    ga[j] = (unsigned long long)(const void*)(A + aoff + (long long)(m0 + r) * lda + c);
    la[0][j] = lds_addr32(&As[0][r * LDP + c]);
    la[1][j] = lds_addr32(&As[1][r * LDP + c]);
  }
#pragma unroll
  for (int j = 0; j < NB; ++j) {
    int i = tid + j * 256;
    int r = i >> 2, c = (i & 3) * 8;
    gb[j] = (unsigned long long)(const void*)(Bt + boff + (long long)(n0 + r) * ldb + c);
    lb[0][j] = lds_addr32(&Bs[0][r * LDP + c]);
    lb[1][j] = lds_addr32(&Bs[1][r * LDP + c]);
  }

  auto issue = [&](int buf) {
#pragma unroll
    for (int j = 0; j < NA; ++j) { async_ld_b128(la[buf][j], ga[j]); ga[j] += 64; }
#pragma unroll
    for (int j = 0; j < NB; ++j) { async_ld_b128(lb[buf][j], gb[j]); gb[j] += 64; }
  };

  v8f acc[2][2] = {};
  const int nsteps = K >> 5;

  issue(0);  // prologue: fill buffer 0

  for (int s = 0; s < nsteps; ++s) {
    const int cur = s & 1;
    // All waves done reading buffer cur^1 (last used at step s-1) before refill.
    __syncthreads();
    if (s + 1 < nsteps) {
      issue(cur ^ 1);       // overlap next tile's global traffic with compute
      wait_async(NA + NB);  // own writes to buffer `cur` complete (in-order)
    } else {
      wait_async(0);
    }
    __syncthreads();        // everyone's writes to buffer `cur` visible

    // Fragment loads per ISA 16-bit A/B layout:
    // lanes 0-15 hold K {0..7, 16..23}; lanes 16-31 hold K {8..15, 24..31}
    Frag16 af[2], bfr[2];
#pragma unroll
    for (int t = 0; t < 2; ++t) {
      int ar = (wm * 32 + t * 16 + l16) * LDP;
      af[t].h[0] = *(const v8bf*)&As[cur][ar + kh * 8];
      af[t].h[1] = *(const v8bf*)&As[cur][ar + 16 + kh * 8];
      int br = (wn * 32 + t * 16 + l16) * LDP;
      bfr[t].h[0] = *(const v8bf*)&Bs[cur][br + kh * 8];
      bfr[t].h[1] = *(const v8bf*)&Bs[cur][br + 16 + kh * 8];
    }
#pragma unroll
    for (int tm = 0; tm < 2; ++tm)
#pragma unroll
      for (int tn = 0; tn < 2; ++tn)
        acc[tm][tn] = __builtin_amdgcn_wmma_f32_16x16x32_bf16(
            false, af[tm].v, false, bfr[tn].v, (short)0, acc[tm][tn],
            false, false);
  }

  // Epilogue: C layout — VGPR r, lane: M = kh*8 + r, N = lane&15
#pragma unroll
  for (int tm = 0; tm < 2; ++tm) {
#pragma unroll
    for (int tn = 0; tn < 2; ++tn) {
      int col = n0 + wn * 32 + tn * 16 + l16;
#pragma unroll
      for (int r = 0; r < 8; ++r) {
        int row = m0 + wm * 32 + tm * 16 + kh * 8 + r;
        float vv = acc[tm][tn][r] * alpha;
        if (bias) vv += bias[col];
        long long ci = coff + (long long)row * ldc + col;
        if (MODE == 0) {
          if (relu) vv = fmaxf(vv, 0.0f);
          ((bf16_t*)Cv)[ci] = (bf16_t)vv;
        } else if (MODE == 1) {
          ((float*)Cv)[ci] = vv;
        } else {
          ((float*)Cv)[ci] = vv + resid[ci];
        }
      }
    }
  }
}

// ---------------------------------------------------------------------------
// LayerNorm over D=512: one block (256 thr) per row, 2 elems/thread.
// ---------------------------------------------------------------------------
template<typename OutT>
__global__ __launch_bounds__(256) void layernorm_kernel(
    const float* __restrict__ x, const float* __restrict__ g,
    const float* __restrict__ b, OutT* __restrict__ out)
{
  __shared__ float red[256];
  const int row = blockIdx.x;
  const int t = threadIdx.x;
  const float* xr = x + (long long)row * DD;
  float v0 = xr[t], v1 = xr[t + 256];
  red[t] = v0 + v1;
  __syncthreads();
  for (int o = 128; o > 0; o >>= 1) {
    if (t < o) red[t] += red[t + o];
    __syncthreads();
  }
  float mean = red[0] * (1.0f / DD);
  __syncthreads();
  float d0 = v0 - mean, d1 = v1 - mean;
  red[t] = d0 * d0 + d1 * d1;
  __syncthreads();
  for (int o = 128; o > 0; o >>= 1) {
    if (t < o) red[t] += red[t + o];
    __syncthreads();
  }
  float inv = rsqrtf(red[0] * (1.0f / DD) + 1e-5f);
  out[(long long)row * DD + t]       = (OutT)(d0 * inv * g[t] + b[t]);
  out[(long long)row * DD + t + 256] = (OutT)(d1 * inv * g[t + 256] + b[t + 256]);
}

// ---------------------------------------------------------------------------
// Masked softmax over key dim (Tk=512): block per (tq, h, b), writes bf16 probs
// ---------------------------------------------------------------------------
__global__ __launch_bounds__(256) void softmax_kernel(
    const float* __restrict__ scores, const int* __restrict__ toks,
    bf16_t* __restrict__ probs, int causal)
{
  __shared__ float red[256];
  const int tq = blockIdx.x, h = blockIdx.y, b = blockIdx.z;
  const long long base = ((long long)(b * HH + h) * TT + tq) * TT;
  const int t = threadIdx.x;
  const int s0 = t, s1 = t + 256;
  bool ok0 = (toks[b * TT + s0] != 0) && (!causal || s0 <= tq);
  bool ok1 = (toks[b * TT + s1] != 0) && (!causal || s1 <= tq);
  float v0 = ok0 ? scores[base + s0] : -1e30f;
  float v1 = ok1 ? scores[base + s1] : -1e30f;
  red[t] = fmaxf(v0, v1);
  __syncthreads();
  for (int o = 128; o > 0; o >>= 1) {
    if (t < o) red[t] = fmaxf(red[t], red[t + o]);
    __syncthreads();
  }
  float mx = red[0];
  __syncthreads();
  float e0 = ok0 ? __expf(v0 - mx) : 0.0f;
  float e1 = ok1 ? __expf(v1 - mx) : 0.0f;
  red[t] = e0 + e1;
  __syncthreads();
  for (int o = 128; o > 0; o >>= 1) {
    if (t < o) red[t] += red[t + o];
    __syncthreads();
  }
  float inv = 1.0f / red[0];
  probs[base + s0] = (bf16_t)(e0 * inv);
  probs[base + s1] = (bf16_t)(e1 * inv);
}

// ---------------------------------------------------------------------------
// Embedding gather + positional add (fp32 residual stream init)
// ---------------------------------------------------------------------------
__global__ __launch_bounds__(256) void embed_kernel(
    const int* __restrict__ toks, const float* __restrict__ tok_emb,
    const float* __restrict__ pos_emb, float* __restrict__ x)
{
  const int bt = blockIdx.x;
  const int tpos = bt % TT;
  const int id = toks[bt];
  const int t = threadIdx.x;
  x[(long long)bt * DD + t] =
      tok_emb[(long long)id * DD + t] + pos_emb[(long long)tpos * DD + t];
  x[(long long)bt * DD + t + 256] =
      tok_emb[(long long)id * DD + t + 256] + pos_emb[(long long)tpos * DD + t + 256];
}

// V (b,s,h,e) -> Vt (b,h,e,s)  (bf16)
__global__ __launch_bounds__(256) void transpose_v_kernel(
    const bf16_t* __restrict__ V, bf16_t* __restrict__ Vt)
{
  long long i = (long long)blockIdx.x * 256 + threadIdx.x;  // B*T*H*HD = 2^20
  int e = i & 63;
  int h = (i >> 6) & 7;
  int s = (i >> 9) & 511;
  int b = (int)(i >> 18);
  Vt[((long long)(b * HH + h) * HDIM + e) * TT + s] = V[i];
}

// W (KxN f32 row-major) -> Wt (NxK bf16)
__global__ __launch_bounds__(256) void repack_t_kernel(
    const float* __restrict__ W, bf16_t* __restrict__ Wt, int K, int N)
{
  long long i = (long long)blockIdx.x * 256 + threadIdx.x;
  if (i >= (long long)K * N) return;
  int k = (int)(i % K);
  long long n = i / K;
  Wt[i] = (bf16_t)W[(long long)k * N + n];
}

// W (H,D,HD) f32 -> Wt ((H*HD) x D) bf16
__global__ __launch_bounds__(256) void repack_heads_kernel(
    const float* __restrict__ W, bf16_t* __restrict__ Wt)
{
  long long i = (long long)blockIdx.x * 256 + threadIdx.x;  // 262144
  int d = i & 511;
  int ne = (int)(i >> 9);
  int e = ne & 63;
  int h = ne >> 6;
  Wt[i] = (bf16_t)W[((long long)h * DD + d) * HDIM + e];
}

// ---------------------------------------------------------------------------
extern "C" void kernel_launch(void* const* d_in, const int* in_sizes, int n_in,
                              void* d_out, int out_size, void* d_ws, size_t ws_size,
                              hipStream_t stream)
{
  (void)in_sizes; (void)n_in; (void)out_size; (void)ws_size;
  auto F = [&](int i) { return (const float*)d_in[i]; };
  auto I = [&](int i) { return (const int*)d_in[i]; };

  const int* src = I(0);
  const int* tgt = I(1);
  const float* src_tok = F(2);
  const float* src_pos = F(3);
  const float* tgt_tok = F(4);
  const float* tgt_pos = F(5);
  const int ENC0 = 6;
  const int DEC0 = ENC0 + 13 * LL;          // 84
  const int POST = DEC0 + 22 * LL;          // 216
  const float* enc_ln_g = F(POST + 0); const float* enc_ln_b = F(POST + 1);
  const float* dec_ln_g = F(POST + 2); const float* dec_ln_b = F(POST + 3);
  const float* head_w   = F(POST + 4); const float* head_b   = F(POST + 5);

  // --- workspace bump allocator ---
  char* ws = (char*)d_ws;
  size_t off = 0;
  auto alloc = [&](size_t bytes) -> void* {
    void* p = ws + off;
    off = (off + bytes + 255) & ~(size_t)255;
    return p;
  };
  auto allocB = [&](size_t n) { return (bf16_t*)alloc(n * sizeof(bf16_t)); };
  auto allocF = [&](size_t n) { return (float*)alloc(n * sizeof(float)); };

  bf16_t *ewq[LL], *ewk[LL], *ewv[LL], *ewo[LL], *ew1[LL], *ew2[LL];
  bf16_t *dswq[LL], *dswk[LL], *dswv[LL], *dswo[LL];
  bf16_t *dcwq[LL], *dcwk[LL], *dcwv[LL], *dcwo[LL];
  bf16_t *dw1[LL], *dw2[LL];
  for (int l = 0; l < LL; ++l) {
    ewq[l] = allocB((size_t)DD * DD); ewk[l] = allocB((size_t)DD * DD);
    ewv[l] = allocB((size_t)DD * DD); ewo[l] = allocB((size_t)DD * DD);
    ew1[l] = allocB((size_t)DD * 4 * DD); ew2[l] = allocB((size_t)DD * 4 * DD);
    dswq[l] = allocB((size_t)DD * DD); dswk[l] = allocB((size_t)DD * DD);
    dswv[l] = allocB((size_t)DD * DD); dswo[l] = allocB((size_t)DD * DD);
    dcwq[l] = allocB((size_t)DD * DD); dcwk[l] = allocB((size_t)DD * DD);
    dcwv[l] = allocB((size_t)DD * DD); dcwo[l] = allocB((size_t)DD * DD);
    dw1[l] = allocB((size_t)DD * 4 * DD); dw2[l] = allocB((size_t)DD * 4 * DD);
  }
  bf16_t* headt = allocB((size_t)VV * DD);

  const size_t MTOT = (size_t)BB * TT;  // 2048 rows
  float*  xb   = allocF(MTOT * DD);
  float*  yb   = allocF(MTOT * DD);
  float*  encb = allocF(MTOT * DD);
  bf16_t* hb   = allocB(MTOT * DD);
  bf16_t* eb   = allocB(MTOT * DD);
  bf16_t* qb   = allocB(MTOT * DD);
  bf16_t* kb   = allocB(MTOT * DD);
  bf16_t* vb   = allocB(MTOT * DD);
  bf16_t* vtb  = allocB(MTOT * DD);
  bf16_t* aob  = allocB(MTOT * DD);
  bf16_t* midb = allocB(MTOT * 4 * (size_t)DD);
  float*  scoresb = allocF((size_t)BB * HH * TT * TT);
  bf16_t* probsb  = allocB((size_t)BB * HH * TT * TT);

  // --- weight repack fp32 -> bf16, transposed to (N x K) ---
  auto rpH = [&](const float* W, bf16_t* Wt) {
    repack_heads_kernel<<<(HH * DD * HDIM) / 256, 256, 0, stream>>>(W, Wt);
  };
  auto rpT = [&](const float* W, bf16_t* Wt, int K, int N) {
    repack_t_kernel<<<(unsigned)(((long long)K * N + 255) / 256), 256, 0, stream>>>(W, Wt, K, N);
  };
  for (int l = 0; l < LL; ++l) {
    int b = ENC0 + 13 * l;
    rpH(F(b + 0), ewq[l]); rpH(F(b + 1), ewk[l]); rpH(F(b + 2), ewv[l]);
    rpT(F(b + 3), ewo[l], DD, DD);
    rpT(F(b + 5), ew1[l], DD, 4 * DD);
    rpT(F(b + 7), ew2[l], 4 * DD, DD);
    int d = DEC0 + 22 * l;
    rpH(F(d + 0), dswq[l]); rpH(F(d + 1), dswk[l]); rpH(F(d + 2), dswv[l]);
    rpT(F(d + 3), dswo[l], DD, DD);
    rpH(F(d + 5), dcwq[l]); rpH(F(d + 6), dcwk[l]); rpH(F(d + 7), dcwv[l]);
    rpT(F(d + 8), dcwo[l], DD, DD);
    rpT(F(d + 10), dw1[l], DD, 4 * DD);
    rpT(F(d + 12), dw2[l], 4 * DD, DD);
  }
  rpT(head_w, headt, DD, VV);

  // --- GEMM helpers ---
  auto gemmB = [&](const bf16_t* A, const bf16_t* Wt, bf16_t* C,
                   const float* bias, int relu, int N, int K) {
    dim3 g(N / 128, (unsigned)(MTOT / 64), 1);
    gemm_bf16_kernel<2, 4, 0><<<g, 256, 0, stream>>>(
        A, K, 0, 0, Wt, K, 0, 0, C, N, 0, 0, nullptr, bias, 1, K, 1.0f, relu);
  };
  auto gemmRes = [&](const bf16_t* A, const bf16_t* Wt, float* xres,
                     const float* bias, int K) {
    dim3 g(DD / 128, (unsigned)(MTOT / 64), 1);
    gemm_bf16_kernel<2, 4, 2><<<g, 256, 0, stream>>>(
        A, K, 0, 0, Wt, K, 0, 0, xres, DD, 0, 0, xres, bias, 1, K, 1.0f, 0);
  };

  auto mha = [&](const bf16_t* hq, const bf16_t* hkv,
                 const bf16_t* wq, const bf16_t* wk, const bf16_t* wv,
                 const bf16_t* wo, const float* bo,
                 const int* mtoks, int causal, float* xres) {
    gemmB(hq,  wq, qb, nullptr, 0, DD, DD);
    gemmB(hkv, wk, kb, nullptr, 0, DD, DD);
    gemmB(hkv, wv, vb, nullptr, 0, DD, DD);
    transpose_v_kernel<<<(unsigned)(MTOT * DD / 256), 256, 0, stream>>>(vb, vtb);
    {  // scores = (1/8) Q K^T   batched over (b,h)
      dim3 g(TT / 128, TT / 64, BB * HH);
      gemm_bf16_kernel<2, 4, 1><<<g, 256, 0, stream>>>(
          qb, DD, (long long)TT * DD, HDIM,
          kb, DD, (long long)TT * DD, HDIM,
          scoresb, TT, (long long)HH * TT * TT, (long long)TT * TT,
          nullptr, nullptr, HH, HDIM, 0.125f, 0);
    }
    softmax_kernel<<<dim3(TT, HH, BB), 256, 0, stream>>>(scoresb, mtoks, probsb, causal);
    {  // out = P V   batched over (b,h), N = 64 -> 128x64 tiles
      dim3 g(HDIM / 64, TT / 128, BB * HH);
      gemm_bf16_kernel<4, 2, 0><<<g, 256, 0, stream>>>(
          probsb, TT, (long long)HH * TT * TT, (long long)TT * TT,
          vtb, TT, (long long)HH * HDIM * TT, (long long)HDIM * TT,
          aob, DD, (long long)TT * DD, HDIM,
          nullptr, nullptr, HH, TT, 1.0f, 0);
    }
    gemmRes(aob, wo, xres, bo, DD);  // x += attn @ wo + bo
  };

  auto ffn = [&](const bf16_t* h, const bf16_t* w1, const float* b1,
                 const bf16_t* w2, const float* b2, float* xres) {
    gemmB(h, w1, midb, b1, 1, 4 * DD, DD);   // relu(h@w1+b1) bf16
    gemmRes(midb, w2, xres, b2, 4 * DD);     // x += mid@w2+b2
  };

  auto lnB = [&](const float* x, const float* g, const float* b, bf16_t* o) {
    layernorm_kernel<bf16_t><<<(unsigned)MTOT, 256, 0, stream>>>(x, g, b, o);
  };
  auto lnF = [&](const float* x, const float* g, const float* b, float* o) {
    layernorm_kernel<float><<<(unsigned)MTOT, 256, 0, stream>>>(x, g, b, o);
  };

  // ----- encoder -----
  embed_kernel<<<(unsigned)MTOT, 256, 0, stream>>>(src, src_tok, src_pos, xb);
  for (int l = 0; l < LL; ++l) {
    int b = ENC0 + 13 * l;
    lnB(xb, F(b + 9), F(b + 10), hb);
    mha(hb, hb, ewq[l], ewk[l], ewv[l], ewo[l], F(b + 4), src, 0, xb);
    lnB(xb, F(b + 11), F(b + 12), hb);
    ffn(hb, ew1[l], F(b + 6), ew2[l], F(b + 8), xb);
  }
  lnF(xb, enc_ln_g, enc_ln_b, encb);

  // ----- decoder -----
  embed_kernel<<<(unsigned)MTOT, 256, 0, stream>>>(tgt, tgt_tok, tgt_pos, yb);
  for (int l = 0; l < LL; ++l) {
    int d = DEC0 + 22 * l;
    lnB(yb, F(d + 14), F(d + 15), hb);
    mha(hb, hb, dswq[l], dswk[l], dswv[l], dswo[l], F(d + 4), tgt, 1, yb);
    lnB(encb, F(d + 20), F(d + 21), eb);
    lnB(yb, F(d + 16), F(d + 17), hb);
    mha(hb, eb, dcwq[l], dcwk[l], dcwv[l], dcwo[l], F(d + 9), src, 0, yb);
    lnB(yb, F(d + 18), F(d + 19), hb);
    ffn(hb, dw1[l], F(d + 11), dw2[l], F(d + 13), yb);
  }
  lnB(yb, dec_ln_g, dec_ln_b, hb);

  // ----- vocab head: (2048 x 512) @ (512 x 32000) + bias -> fp32 d_out -----
  {
    dim3 g(VV / 128, (unsigned)(MTOT / 64), 1);
    gemm_bf16_kernel<2, 4, 1><<<g, 256, 0, stream>>>(
        hb, DD, 0, 0, headt, DD, 0, 0,
        (float*)d_out, VV, 0, 0, nullptr, head_b, 1, DD, 1.0f, 0);
  }
}
